// Block_8237747274098
// MI455X (gfx1250) — matrix operand order repaired and checked by reading, not compiled, not run on hardware
//
#include <hip/hip_runtime.h>
#include <hip/hip_bf16.h>

// ---------------------------------------------------------------------------
// GPT block for MI455X (gfx1250).
// - All six GEMMs (qkv, q@k^T, att@v, proj, fc, fc2) via V_WMMA_F32_16X16X32_F16
//   (f16 operands / f32 accumulate), fused bias/scale/gelu/residual epilogues.
// - Tile staging uses CDNA5 GLOBAL_LOAD_ASYNC_TO_LDS_B128 (ASYNCcnt-tracked,
//   no VGPR round trip) with explicit s_wait_asynccnt before the barrier.
// - All B operands are pre-transposed to [N,K] so every tile copy is
//   contiguous 16B-per-lane (async-copy friendly, no LDS scatter).
// - Causal structure exploited twice: block-skip in q@k^T, K-limit in att@v.
// ---------------------------------------------------------------------------

typedef _Float16 v16h __attribute__((ext_vector_type(16)));
typedef _Float16 v8h  __attribute__((ext_vector_type(8)));
typedef float    v8f  __attribute__((ext_vector_type(8)));

#define BM 128
#define BN 128
#define BK 32

#define FLAG_GELU    1
#define FLAG_CAUSAL  2   // skip output blocks strictly above the diagonal
#define FLAG_KCAUSAL 4   // A columns (K dim) beyond rowBase+BM are all zero

__device__ __forceinline__ float gelu_tanh(float x) {
    const float c = 0.7978845608028654f;
    float x3 = x * x * x;
    return 0.5f * x * (1.0f + tanhf(c * (x + 0.044715f * x3)));
}

// 16-byte async copy global -> LDS (per active lane), tracked by ASYNCcnt.
__device__ __forceinline__ void async_b128(unsigned lds_off, const _Float16* g) {
    asm volatile("global_load_async_to_lds_b128 %0, %1, off"
                 :: "v"(lds_off), "v"((unsigned long long)g)
                 : "memory");
}
__device__ __forceinline__ void wait_async0() {
    asm volatile("s_wait_asynccnt 0" ::: "memory");
}
__device__ __forceinline__ unsigned lds_off_u32(const void* p) {
    return (unsigned)(unsigned long long)p;   // low 32 bits = LDS byte offset
}

// C[M,N] = epilogue( alpha * A[M,K] @ B[N,K]^T + bias ) (+resid), out f32/f16.
// A: f16 row-major [M,K], lda.  B: f16 row-major [N,K], ldb.
// Batched via blockIdx.z with element strides sA/sB/sC. M,N%128==0, K%32==0.
__global__ void __launch_bounds__(256, 2)
wmma_gemm_f16(const _Float16* __restrict__ A, const _Float16* __restrict__ B,
              const float* __restrict__ bias, const float* __restrict__ resid,
              float* __restrict__ outF, _Float16* __restrict__ outH,
              int M, int N, int K, int lda, int ldb, int ldc,
              long long sA, long long sB, long long sC,
              float alpha, int flags)
{
    __shared__ _Float16 As[2][BM * BK];
    __shared__ _Float16 Bs[2][BN * BK];

    const int tid   = threadIdx.x;
    const int lane  = tid & 31;
    const int wave  = tid >> 5;      // 8 waves
    const int waveM = wave & 3;      // 4 along M -> 32 rows each
    const int waveN = wave >> 2;     // 2 along N -> 64 cols each

    const int rowBase = blockIdx.y * BM;
    const int colBase = blockIdx.x * BN;

    if ((flags & FLAG_CAUSAL) && (rowBase + BM <= colBase))
        return;  // fully masked; softmax never uses these scores

    int Kend = K;
    if (flags & FLAG_KCAUSAL) {      // att rows in this block are zero past diag
        const int kl = rowBase + BM;
        Kend = (kl < K) ? kl : K;
    }

    A += (long long)blockIdx.z * sA + (long long)rowBase * lda;
    B += (long long)blockIdx.z * sB + (long long)colBase * ldb;
    const long long cOff = (long long)blockIdx.z * sC;

    v8f acc[2][4];
    #pragma unroll
    for (int mi = 0; mi < 2; ++mi)
        #pragma unroll
        for (int ni = 0; ni < 4; ++ni)
            #pragma unroll
            for (int r = 0; r < 8; ++r)
                acc[mi][ni][r] = 0.0f;

    // Cooperative async staging of one (128 x 32) A tile and B tile.
    // 512 chunks of 8 halves each per tile; 256 threads -> 2 chunks each.
    auto copy_tiles = [&](int buf, int k0) {
        #pragma unroll
        for (int i = 0; i < 2; ++i) {
            const int chunk = tid + i * 256;
            const int r = chunk >> 2;            // BK/8 = 4 chunks per row
            const int c = (chunk & 3) << 3;
            async_b128(lds_off_u32(&As[buf][r * BK + c]),
                       A + (long long)r * lda + (k0 + c));
            async_b128(lds_off_u32(&Bs[buf][r * BK + c]),
                       B + (long long)r * ldb + (k0 + c));
        }
    };

    const int lrow = lane & 15;
    const int kbA  = (lane & 16) ? 16 : 0;   // A layout: K 0-15 vs 16-31
    const int kbB  = (lane & 16) ? 16 : 0;   // B layout: K 0-15 vs 16-31

    int buf = 0;
    copy_tiles(0, 0);
    wait_async0();
    __syncthreads();

    for (int k0 = 0; k0 < Kend; k0 += BK) {
        const int nbuf = buf ^ 1;
        if (k0 + BK < Kend) copy_tiles(nbuf, k0 + BK);

        v16h afr[2];
        v16h bfr[4];
        #pragma unroll
        for (int mi = 0; mi < 2; ++mi) {
            const _Float16* p = &As[buf][(waveM * 32 + mi * 16 + lrow) * BK];
            v8h lo = *(const v8h*)(p + kbA);        // K = kbA .. kbA+7
            v8h hi = *(const v8h*)(p + kbA + 8);    // K = kbA+8 .. kbA+15
            afr[mi] = __builtin_shufflevector(lo, hi,
                       0,1,2,3,4,5,6,7,8,9,10,11,12,13,14,15);
        }
        #pragma unroll
        for (int ni = 0; ni < 4; ++ni) {
            const _Float16* p = &Bs[buf][(waveN * 64 + ni * 16 + lrow) * BK + kbB];
            bfr[ni] = *(const v16h*)p;
        }

        #pragma unroll
        for (int mi = 0; mi < 2; ++mi)
            #pragma unroll
            for (int ni = 0; ni < 4; ++ni)
                acc[mi][ni] = __builtin_amdgcn_wmma_f32_16x16x32_f16(
                    false, afr[mi], false, bfr[ni],
                    (short)0, acc[mi][ni], false, false);

        wait_async0();     // async LDS writes landed before cross-wave barrier
        __syncthreads();
        buf = nbuf;
    }

    // Epilogue per C/D layout: VGPR r -> M = r (+8 for lanes 16-31), N = lane&15
    const int rsel = (lane & 16) ? 8 : 0;
    #pragma unroll
    for (int mi = 0; mi < 2; ++mi) {
        #pragma unroll
        for (int ni = 0; ni < 4; ++ni) {
            const int col = colBase + waveN * 64 + ni * 16 + lrow;
            const float bv = bias ? bias[col] : 0.0f;
            #pragma unroll
            for (int r = 0; r < 8; ++r) {
                const int row = rowBase + waveM * 32 + mi * 16 + rsel + r;
                float v = acc[mi][ni][r] * alpha + bv;
                if (flags & FLAG_GELU) v = gelu_tanh(v);
                const long long idx = cOff + (long long)row * ldc + col;
                if (resid) v += resid[idx];
                if (outF) outF[idx] = v;
                if (outH) outH[idx] = (_Float16)v;
            }
        }
    }
}

// LayerNorm over D=1024, one row per block (256 threads x 4 elems), f16 out.
__global__ void __launch_bounds__(256)
layernorm_f16(const float* __restrict__ x, const float* __restrict__ g,
              const float* __restrict__ bb, _Float16* __restrict__ out)
{
    const int D = 1024;
    const long long row = blockIdx.x;
    const float* xr = x + row * D;
    const int tid = threadIdx.x;
    __shared__ float red[256];

    float4 v = ((const float4*)xr)[tid];
    red[tid] = v.x + v.y + v.z + v.w;
    __syncthreads();
    for (int o = 128; o > 0; o >>= 1) {
        if (tid < o) red[tid] += red[tid + o];
        __syncthreads();
    }
    const float mu = red[0] * (1.0f / D);
    __syncthreads();

    const float dx = v.x - mu, dy = v.y - mu, dz = v.z - mu, dw = v.w - mu;
    red[tid] = dx*dx + dy*dy + dz*dz + dw*dw;
    __syncthreads();
    for (int o = 128; o > 0; o >>= 1) {
        if (tid < o) red[tid] += red[tid + o];
        __syncthreads();
    }
    const float rstd = rsqrtf(red[0] * (1.0f / D) + 1e-5f);

    float4 gv = ((const float4*)g)[tid];
    float4 bv = ((const float4*)bb)[tid];
    _Float16* orow = out + row * D;
    orow[tid*4+0] = (_Float16)(dx * rstd * gv.x + bv.x);
    orow[tid*4+1] = (_Float16)(dy * rstd * gv.y + bv.y);
    orow[tid*4+2] = (_Float16)(dz * rstd * gv.z + bv.z);
    orow[tid*4+3] = (_Float16)(dw * rstd * gv.w + bv.w);
}

// Causal softmax over row t (length T=2048), f32 scores in, f16 probs out.
__global__ void __launch_bounds__(256)
softmax_causal(const float* __restrict__ sc, _Float16* __restrict__ att)
{
    const int T = 2048;
    const int t = blockIdx.x;
    const long long base = ((long long)blockIdx.y * T + t) * T;
    const float* row = sc + base;
    _Float16* orow = att + base;
    const int tid = threadIdx.x;
    __shared__ float red[256];

    float vals[8];
    float m = -3.0e38f;
    #pragma unroll
    for (int i = 0; i < 8; ++i) {
        const int s = tid * 8 + i;
        const float v = row[s];
        vals[i] = v;
        if (s <= t && v > m) m = v;
    }
    red[tid] = m;
    __syncthreads();
    for (int o = 128; o > 0; o >>= 1) {
        if (tid < o) red[tid] = fmaxf(red[tid], red[tid + o]);
        __syncthreads();
    }
    m = red[0];
    __syncthreads();

    float sum = 0.0f;
    #pragma unroll
    for (int i = 0; i < 8; ++i) {
        const int s = tid * 8 + i;
        const float e = (s <= t) ? __expf(vals[i] - m) : 0.0f;
        vals[i] = e;
        sum += e;
    }
    red[tid] = sum;
    __syncthreads();
    for (int o = 128; o > 0; o >>= 1) {
        if (tid < o) red[tid] += red[tid + o];
        __syncthreads();
    }
    const float inv = 1.0f / red[0];
    #pragma unroll
    for (int i = 0; i < 8; ++i)
        orow[tid * 8 + i] = (_Float16)(vals[i] * inv);
}

// Transpose [R,C] f32 -> [C,R] f16 (fused convert). 32x32 tiles, 256 threads.
__global__ void __launch_bounds__(256)
transpose_f32_to_f16(const float* __restrict__ in, _Float16* __restrict__ out,
                     int R, int C, int ldi, int ldo)
{
    __shared__ float tile[32][33];
    const int r0 = blockIdx.y * 32, c0 = blockIdx.x * 32;
    const int tx = threadIdx.x & 31, ty = threadIdx.x >> 5;
    #pragma unroll
    for (int i = 0; i < 4; ++i)
        tile[ty + i * 8][tx] = in[(long long)(r0 + ty + i * 8) * ldi + c0 + tx];
    __syncthreads();
    #pragma unroll
    for (int i = 0; i < 4; ++i)
        out[(long long)(c0 + ty + i * 8) * ldo + r0 + tx] =
            (_Float16)tile[tx][ty + i * 8];
}

// Transpose [R,C] f16 -> [C,R] f16 (batched, for v -> v^T).
__global__ void __launch_bounds__(256)
transpose_f16(const _Float16* __restrict__ in, _Float16* __restrict__ out,
              int R, int C, int ldi, int ldo, long long sIn, long long sOut)
{
    __shared__ _Float16 tile[32][34];
    in  += (long long)blockIdx.z * sIn;
    out += (long long)blockIdx.z * sOut;
    const int r0 = blockIdx.y * 32, c0 = blockIdx.x * 32;
    const int tx = threadIdx.x & 31, ty = threadIdx.x >> 5;
    #pragma unroll
    for (int i = 0; i < 4; ++i)
        tile[ty + i * 8][tx] = in[(long long)(r0 + ty + i * 8) * ldi + c0 + tx];
    __syncthreads();
    #pragma unroll
    for (int i = 0; i < 4; ++i)
        out[(long long)(c0 + ty + i * 8) * ldo + r0 + tx] = tile[tx][ty + i * 8];
}

extern "C" void kernel_launch(void* const* d_in, const int* in_sizes, int n_in,
                              void* d_out, int out_size, void* d_ws, size_t ws_size,
                              hipStream_t stream)
{
    (void)in_sizes; (void)n_in; (void)out_size; (void)ws_size;

    const float* x      = (const float*)d_in[0];
    const float* w_attn = (const float*)d_in[1];
    const float* b_attn = (const float*)d_in[2];
    const float* w_proj = (const float*)d_in[3];
    const float* b_proj = (const float*)d_in[4];
    const float* ln1_g  = (const float*)d_in[5];
    const float* ln1_b  = (const float*)d_in[6];
    const float* ln2_g  = (const float*)d_in[7];
    const float* ln2_b  = (const float*)d_in[8];
    const float* w_fc   = (const float*)d_in[9];
    const float* b_fc   = (const float*)d_in[10];
    const float* w_fc2  = (const float*)d_in[11];
    const float* b_fc2  = (const float*)d_in[12];
    float* out = (float*)d_out;

    const int Bb = 4, T = 2048, D = 1024;
    const int M = Bb * T;  // 8192

    // workspace carve-up (256B aligned)
    char* ws = (char*)d_ws;
    size_t off = 0;
    auto take = [&](size_t bytes) -> void* {
        void* p = ws + off;
        off += (bytes + 255) & ~(size_t)255;
        return p;
    };
    _Float16* slotA    = (_Float16*)take((size_t)M * D * 2);       // h16 -> y16 -> h2_16
    _Float16* wattnT   = (_Float16*)take((size_t)3 * D * D * 2);   // [3D, D]
    _Float16* wprojT   = (_Float16*)take((size_t)D * D * 2);       // [D, D]
    _Float16* wfcT     = (_Float16*)take((size_t)4 * D * D * 2);   // [4D, D]
    _Float16* wfc2T    = (_Float16*)take((size_t)D * 4 * D * 2);   // [D, 4D]
    _Float16* qkv16    = (_Float16*)take((size_t)M * 3 * D * 2);
    _Float16* vT16     = (_Float16*)take((size_t)Bb * D * T * 2);  // [D, T] per batch
    float*    scores   = (float*)   take((size_t)Bb * T * T * 4);  // reused as afc16
    _Float16* att16    = (_Float16*)take((size_t)Bb * T * T * 2);
    float*    x1       = (float*)   take((size_t)M * D * 4);
    _Float16* afc16    = (_Float16*)scores;

    // weights: f32 [K,N] -> f16 [N,K] (fused transpose+convert)
    transpose_f32_to_f16<<<dim3(3 * D / 32, D / 32), 256, 0, stream>>>(
        w_attn, wattnT, D, 3 * D, 3 * D, D);
    transpose_f32_to_f16<<<dim3(D / 32, D / 32), 256, 0, stream>>>(
        w_proj, wprojT, D, D, D, D);
    transpose_f32_to_f16<<<dim3(4 * D / 32, D / 32), 256, 0, stream>>>(
        w_fc, wfcT, D, 4 * D, 4 * D, D);
    transpose_f32_to_f16<<<dim3(D / 32, 4 * D / 32), 256, 0, stream>>>(
        w_fc2, wfc2T, 4 * D, D, D, 4 * D);

    // h = LN1(x)  (f16)
    layernorm_f16<<<M, 256, 0, stream>>>(x, ln1_g, ln1_b, slotA);

    // qkv = h @ w_attn + b_attn  -> f16 [M, 3D]
    wmma_gemm_f16<<<dim3(3 * D / BN, M / BM, 1), 256, 0, stream>>>(
        slotA, wattnT, b_attn, nullptr, nullptr, qkv16,
        M, 3 * D, D, D, D, 3 * D, 0, 0, 0, 1.0f, 0);

    // v^T per batch (for att@v as A[M,K] @ B[N,K]^T)
    transpose_f16<<<dim3(D / 32, T / 32, Bb), 256, 0, stream>>>(
        qkv16 + 2 * D, vT16, T, D, 3 * D, T,
        (long long)T * 3 * D, (long long)D * T);

    // scores = q @ k^T * (1/sqrt(64))  (batched, causal block-skip)
    wmma_gemm_f16<<<dim3(T / BN, T / BM, Bb), 256, 0, stream>>>(
        qkv16, qkv16 + D, nullptr, nullptr, scores, nullptr,
        T, T, D, 3 * D, 3 * D, T,
        (long long)T * 3 * D, (long long)T * 3 * D, (long long)T * T,
        0.125f, FLAG_CAUSAL);

    // att = causal softmax(scores) -> f16
    softmax_causal<<<dim3(T, Bb), 256, 0, stream>>>(scores, att16);

    // y = att @ v  -> f16 (into slotA); K limited to rowBase+BM (causal zeros)
    wmma_gemm_f16<<<dim3(D / BN, T / BM, Bb), 256, 0, stream>>>(
        att16, vT16, nullptr, nullptr, nullptr, slotA,
        T, D, T, T, T, D,
        (long long)T * T, (long long)D * T, (long long)T * D,
        1.0f, FLAG_KCAUSAL);

    // x1 = x + y @ w_proj + b_proj  (f32)
    wmma_gemm_f16<<<dim3(D / BN, M / BM, 1), 256, 0, stream>>>(
        slotA, wprojT, b_proj, x, x1, nullptr,
        M, D, D, D, D, D, 0, 0, 0, 1.0f, 0);

    // h2 = LN2(x1)  (f16, into slotA)
    layernorm_f16<<<M, 256, 0, stream>>>(x1, ln2_g, ln2_b, slotA);

    // afc = gelu(h2 @ w_fc + b_fc)  -> f16
    wmma_gemm_f16<<<dim3(4 * D / BN, M / BM, 1), 256, 0, stream>>>(
        slotA, wfcT, b_fc, nullptr, nullptr, afc16,
        M, 4 * D, D, D, D, 4 * D, 0, 0, 0, 1.0f, FLAG_GELU);

    // out = x1 + afc @ w_fc2 + b_fc2  (f32)
    wmma_gemm_f16<<<dim3(D / BN, M / BM, 1), 256, 0, stream>>>(
        afc16, wfc2T, b_fc2, x1, out, nullptr,
        M, D, 4 * D, 4 * D, 4 * D, D, 0, 0, 0, 1.0f, 0);
}
